// Connector_76141180224098
// MI455X (gfx1250) — matrix operand order repaired and checked by reading, not compiled, not run on hardware
//
#include <hip/hip_runtime.h>
#include <hip/hip_bf16.h>
#include <math.h>

// ---------------- problem constants (fixed by the reference) ----------------
#define BATCH 8
#define SEQ   2048
#define DIM   1024   // D  (N of GEMM)
#define DVIS  1152   // DV (K of GEMM)

// ---------------- WMMA types ----------------
typedef __attribute__((ext_vector_type(16))) __bf16 v16bf;
typedef __attribute__((ext_vector_type(8)))  float  v8f;

// f32 -> bf16 round-to-nearest-even (packed pair)
__device__ __forceinline__ unsigned short f2bf(float f) {
    unsigned u = __builtin_bit_cast(unsigned, f);
    unsigned r = u + 0x7FFFu + ((u >> 16) & 1u);
    return (unsigned short)(r >> 16);
}
__device__ __forceinline__ unsigned pk2(float a, float b) {
    return (unsigned)f2bf(a) | ((unsigned)f2bf(b) << 16);
}

// ============================================================================
// Kernel 1: projection GEMM  C[M,N] = A[M,K] * W[K,N] + bias
//   M = B*Nv (18432), K = DVIS (1152), N = DIM (1024)
//   Block tile 128x128, K-step 32, 8 waves of 32 (wave32), wave tile 32x64.
//   v_wmma_f32_16x16x32_bf16, f32 accumulation.
// ============================================================================
#define TILE_K 32
#define LDR    40          // bf16 elements per LDS row (80 B: 16B-aligned, bank-spread)

__global__ __launch_bounds__(256)
void gemm_proj_kernel(const float* __restrict__ A,   // [M, K] visual features
                      const float* __restrict__ W,   // [K, N] projection
                      const float* __restrict__ bias,// [N]
                      float* __restrict__ C)         // [M, N] projected
{
    constexpr int K = DVIS, N = DIM;
    __shared__ __align__(16) __bf16 sA[128 * LDR];   // A tile, row-major [m][k]
    __shared__ __align__(16) __bf16 sB[128 * LDR];   // W tile, transposed [n][k]

    const int tid   = threadIdx.x;
    const int lane  = tid & 31;
    const int wave  = tid >> 5;
    const int waveM = wave & 3;          // 4 waves along M
    const int waveN = wave >> 2;         // 2 waves along N
    const int l15   = lane & 15;
    const int lh    = lane >> 4;         // lane half (0/1)
    const int bM0   = blockIdx.y * 128;
    const int bN0   = blockIdx.x * 128;

    v8f acc[2][4];
    for (int m = 0; m < 2; ++m)
        for (int n = 0; n < 4; ++n)
            acc[m][n] = (v8f)(0.0f);

    union Frag { v16bf v; uint4 u[2]; };

    for (int k0 = 0; k0 < K; k0 += TILE_K) {
        // ---- A tile: 128 rows x 32 K of f32 -> bf16. 8 lanes cover one row's
        // 32 floats (128B contiguous per 8-lane group -> coalesced).
        {
            const int kk = (tid & 7) * 4;
            for (int p = 0; p < 4; ++p) {
                const int row = p * 32 + (tid >> 3);
                const float4 v = *(const float4*)(A + (long)(bM0 + row) * K + k0 + kk);
                uint2 w; w.x = pk2(v.x, v.y); w.y = pk2(v.z, v.w);
                *(uint2*)(sA + row * LDR + kk) = w;   // 8B store, aligned
            }
        }
        // ---- W tile transposed: thread owns one n column, 16 K values.
        // Per iteration all 128 n-lanes read consecutive addresses (coalesced).
        {
            const int n  = tid & 127;
            const int kb = (tid >> 7) * 16;
            for (int kk = 0; kk < 16; kk += 2) {
                float f0 = W[(long)(k0 + kb + kk)     * N + bN0 + n];
                float f1 = W[(long)(k0 + kb + kk + 1) * N + bN0 + n];
                *(unsigned*)(sB + n * LDR + kb + kk) = pk2(f0, f1);
            }
        }
        __syncthreads();

        // ---- fragment loads (contiguous k = lh*16 + s, same order in A and B) ----
        Frag aF[2], bF[4];
        for (int mt = 0; mt < 2; ++mt) {
            // A 16x32 bf16: lane holds row (l15), K = lh*16 .. lh*16+15 contiguous
            const __bf16* base = sA + (waveM * 32 + mt * 16 + l15) * LDR;
            aF[mt].u[0] = *(const uint4*)(base + lh * 16);       // K = lh*16 .. +7
            aF[mt].u[1] = *(const uint4*)(base + lh * 16 + 8);   // K = lh*16+8 .. +7
        }
        for (int nt = 0; nt < 4; ++nt) {
            // B 32x16 bf16: lane holds column (l15), K = lh*16 .. lh*16+15 contiguous
            const __bf16* base = sB + (waveN * 64 + nt * 16 + l15) * LDR + lh * 16;
            bF[nt].u[0] = ((const uint4*)base)[0];
            bF[nt].u[1] = ((const uint4*)base)[1];
        }

        for (int mt = 0; mt < 2; ++mt)
            for (int nt = 0; nt < 4; ++nt)
                acc[mt][nt] = __builtin_amdgcn_wmma_f32_16x16x32_bf16(
                    false, aF[mt].v, false, bF[nt].v,
                    (short)0, acc[mt][nt], false, false);

        __syncthreads();
    }

    // ---- epilogue: C layout = VGPR r -> M = r + lh*8, lanes 0-15 -> N = l15 ----
    for (int nt = 0; nt < 4; ++nt) {
        const int col = bN0 + waveN * 64 + nt * 16 + l15;
        const float bv = bias[col];
        for (int mt = 0; mt < 2; ++mt) {
            const int rbase = bM0 + waveM * 32 + mt * 16 + lh * 8;
            for (int r = 0; r < 8; ++r)
                C[(long)(rbase + r) * N + col] = acc[mt][nt][r] + bv;
        }
    }
}

// ============================================================================
// Kernel 2: build the ragged plan on device (mirrors the host _plan()).
// One block per batch row. Emits per output row:
//   srcrow: index into [text(0..S-1) ; visual(S..S+Nv-1)], -1 for padding
//   cide:   chunk id (0x7FFFFFFF sentinel for padding)
//   vis:    1 if visual chunk
// ============================================================================
__global__ void plan_kernel(const int* __restrict__ texts, const int* __restrict__ img_id_p,
                            int Nv, int L,
                            int* __restrict__ srcrow, int* __restrict__ cide,
                            int* __restrict__ vis)
{
    const int b = blockIdx.x;
    __shared__ int pos[32];
    __shared__ int npos;
    __shared__ int seg_out[66], seg_src[66], seg_len[66], seg_cid[66], seg_vis[66];
    __shared__ int nseg, ltot;

    if (threadIdx.x == 0) npos = 0;
    __syncthreads();
    const int img_id = *img_id_p;
    for (int i = threadIdx.x; i < SEQ; i += blockDim.x)
        if (texts[(long)b * SEQ + i] == img_id) {
            int o = atomicAdd(&npos, 1);
            if (o < 32) pos[o] = i;
        }
    __syncthreads();

    if (threadIdx.x == 0) {
        int n = npos; if (n > 32) n = 32;
        for (int a = 1; a < n; ++a) {            // insertion sort (n is tiny)
            int v = pos[a], c = a - 1;
            while (c >= 0 && pos[c] > v) { pos[c + 1] = pos[c]; --c; }
            pos[c + 1] = v;
        }
        int ns = 0, out = 0;
        if (n == 0) {
            seg_out[0] = 0; seg_src[0] = 0; seg_len[0] = SEQ;
            seg_cid[0] = 0; seg_vis[0] = 0; ns = 1; out = SEQ;
        } else {
            const int per = Nv / n;
            int tc_s[33], tc_e[33]; int ntc = 0, start = 0;
            for (int i = 0; i < n; ++i) {
                int p = pos[i];
                if (p > start) { tc_s[ntc] = start; tc_e[ntc] = p; ++ntc; }
                start = p + 1;
            }
            if (start < SEQ) { tc_s[ntc] = start; tc_e[ntc] = SEQ; ++ntc; }
            int c = 0;
            for (int i = 0; i < n; ++i) {
                if (i < ntc) {
                    seg_out[ns] = out; seg_src[ns] = tc_s[i];
                    seg_len[ns] = tc_e[i] - tc_s[i];
                    seg_cid[ns] = c; seg_vis[ns] = 0;
                    out += seg_len[ns]; ++ns; ++c;
                }
                seg_out[ns] = out; seg_src[ns] = SEQ + i * per;
                seg_len[ns] = per; seg_cid[ns] = c; seg_vis[ns] = 1;
                out += per; ++ns; ++c;
            }
            if (n < ntc) {
                seg_out[ns] = out; seg_src[ns] = tc_s[ntc - 1];
                seg_len[ns] = tc_e[ntc - 1] - tc_s[ntc - 1];
                seg_cid[ns] = c; seg_vis[ns] = 0;
                out += seg_len[ns]; ++ns;
            }
        }
        nseg = ns; ltot = out;
    }
    __syncthreads();

    for (int s = 0; s < nseg; ++s) {
        const int o = seg_out[s], sr = seg_src[s], ln = seg_len[s];
        const int cd = seg_cid[s], vv = seg_vis[s];
        for (int i = threadIdx.x; i < ln; i += blockDim.x) {
            const long w = (long)b * L + o + i;
            srcrow[w] = sr + i; cide[w] = cd; vis[w] = vv;
        }
    }
    for (int i = ltot + threadIdx.x; i < L; i += blockDim.x) {
        const long w = (long)b * L + i;
        srcrow[w] = -1; cide[w] = 0x7FFFFFFF; vis[w] = 0;
    }
}

// ============================================================================
// Kernel 3: fused gather. One block per output row; 256 threads x float4 = 1024.
// ============================================================================
__global__ __launch_bounds__(256)
void gather_kernel(const float* __restrict__ emb, const float* __restrict__ proj,
                   const int* __restrict__ texts, const int* __restrict__ srcrow,
                   int Nv, int L, float* __restrict__ fused)
{
    const long row = blockIdx.x;                 // b*L + i
    const long b   = row / L;
    const int  src = srcrow[row];
    float4* out4 = (float4*)(fused + row * (long)DIM);
    const int nd4 = DIM / 4;

    if (src < 0) {
        const float4 z = {0.f, 0.f, 0.f, 0.f};
        for (int d = threadIdx.x; d < nd4; d += blockDim.x) out4[d] = z;
        return;
    }
    const float* in;
    if (src < SEQ) {
        const int tok = texts[b * SEQ + src];
        in = emb + (long)tok * DIM;
    } else {
        in = proj + ((long)b * Nv + (src - SEQ)) * (long)DIM;
    }
    const float4* in4 = (const float4*)in;
    for (int d = threadIdx.x; d < nd4; d += blockDim.x) out4[d] = in4[d];
}

// ============================================================================
// Kernel 4: nested block attention mask. One block per (b,i) row; streams L
// floats. cide row stays resident in L2/WGP$, so this is write-BW bound.
// ============================================================================
__global__ __launch_bounds__(256)
void mask_kernel(const int* __restrict__ cide, const int* __restrict__ vis,
                 int L, float* __restrict__ mask)
{
    const long row = blockIdx.x;                 // b*L + i
    const long b   = row / L;
    const long i   = row - b * L;
    float* out = mask + row * (long)L;
    const int ci = cide[row];
    const int nj4 = L >> 2;

    if (ci == 0x7FFFFFFF) {                      // padded i row -> zeros
        const float4 z = {0.f, 0.f, 0.f, 0.f};
        for (int j4 = threadIdx.x; j4 < nj4; j4 += blockDim.x) ((float4*)out)[j4] = z;
        for (int j = (nj4 << 2) + threadIdx.x; j < L; j += blockDim.x) out[j] = 0.f;
        return;
    }
    const int vi = vis[row];
    const int4* crow = (const int4*)(cide + b * (long)L);

    for (int j4 = threadIdx.x; j4 < nj4; j4 += blockDim.x) {
        const int4 cj = crow[j4];
        const long j = (long)j4 << 2;
        float4 m;
        m.x = ((cj.x < ci) || (cj.x == ci && (vi || (j + 0) <= i))) ? 1.f : 0.f;
        m.y = ((cj.y < ci) || (cj.y == ci && (vi || (j + 1) <= i))) ? 1.f : 0.f;
        m.z = ((cj.z < ci) || (cj.z == ci && (vi || (j + 2) <= i))) ? 1.f : 0.f;
        m.w = ((cj.w < ci) || (cj.w == ci && (vi || (j + 3) <= i))) ? 1.f : 0.f;
        ((float4*)out)[j4] = m;
    }
    for (long j = (long)(nj4 << 2) + threadIdx.x; j < L; j += blockDim.x) {
        const int cj = cide[b * (long)L + j];
        out[j] = ((cj < ci) || (cj == ci && (vi || j <= i))) ? 1.f : 0.f;
    }
}

// ============================================================================
extern "C" void kernel_launch(void* const* d_in, const int* in_sizes, int n_in,
                              void* d_out, int out_size, void* d_ws, size_t ws_size,
                              hipStream_t stream)
{
    const float* visual = (const float*)d_in[0];   // [B, Nv, DV]
    const float* emb    = (const float*)d_in[1];   // [V, D]
    const float* W      = (const float*)d_in[2];   // [DV, D]
    const float* bias   = (const float*)d_in[3];   // [D]
    const int*   texts  = (const int*)  d_in[4];   // [B, S]
    const int*   img_id = (const int*)  d_in[5];   // scalar

    const int Nv = in_sizes[0] / (BATCH * DVIS);   // 2304
    const int M  = BATCH * Nv;                     // 18432

    // Solve B*L*D + B*L*L == out_size for L.
    const long long per = (long long)out_size / BATCH;
    long long L = (long long)((-(double)DIM +
                     sqrt((double)DIM * (double)DIM + 4.0 * (double)per)) * 0.5 + 0.5);
    while (L > 0 && L * ((long long)DIM + L) > per) --L;
    while ((L + 1) * ((long long)DIM + L + 1) <= per) ++L;
    const int Li = (int)L;

    // workspace layout
    float* proj   = (float*)d_ws;                              // M*D floats
    char*  p      = (char*)d_ws + (size_t)M * DIM * sizeof(float);
    int*   srcrow = (int*)p;            p += (size_t)BATCH * Li * sizeof(int);
    int*   cide   = (int*)p;            p += (size_t)BATCH * Li * sizeof(int);
    int*   vis    = (int*)p;

    float* fusedOut = (float*)d_out;
    float* maskOut  = fusedOut + (size_t)BATCH * Li * DIM;

    // 1) projection GEMM (WMMA bf16, f32 accumulate)
    dim3 gGrid(DIM / 128, M / 128);
    gemm_proj_kernel<<<gGrid, 256, 0, stream>>>(visual, W, bias, proj);

    // 2) device-side ragged plan
    plan_kernel<<<BATCH, 256, 0, stream>>>(texts, img_id, Nv, Li, srcrow, cide, vis);

    // 3) fused gather
    gather_kernel<<<(unsigned)(BATCH * Li), 256, 0, stream>>>(
        emb, proj, texts, srcrow, Nv, Li, fusedOut);

    // 4) attention mask (bandwidth-dominant)
    mask_kernel<<<(unsigned)(BATCH * Li), 256, 0, stream>>>(cide, vis, Li, maskOut);
}
